// EuclideanCodebook_11166914969822
// MI455X (gfx1250) — matrix operand order repaired and checked by reading, not compiled, not run on hardware
//
#include <hip/hip_runtime.h>
#include <hip/hip_bf16.h>

typedef __attribute__((ext_vector_type(16))) __bf16 v16bf;
typedef __attribute__((ext_vector_type(4)))  __bf16 v4bf;
typedef __attribute__((ext_vector_type(8)))  float  v8f;

#define DECAY            0.99f
#define ONE_MINUS_DECAY  0.01f
#define EPSF             1e-6f

// Problem dims (reference: B=8, S=8192, D=256, K=2048)
constexpr int N = 65536;   // B*S rows
constexpr int D = 256;     // feature dim
constexpr int K = 2048;    // codebook size

// d_out layout (float elements, reference tuple flattened in return order)
constexpr long OFF_Q   = 0;                   // quantize        [N, D]
constexpr long OFF_IND = OFF_Q   + (long)N*D; // embed_ind       [N] (as float)
constexpr long OFF_NCS = OFF_IND + N;         // new_cluster_size[K]
constexpr long OFF_NEA = OFF_NCS + K;         // new_embed_avg   [K, D]
constexpr long OFF_NE  = OFF_NEA + (long)K*D; // new_embed       [K, D]

// d_ws layout (float elements) — tiny: 2049 floats
constexpr long WS_HN    = 0;   // 0.5*||embed_k||^2  [K]
constexpr long WS_TOTAL = K;   // scalar sum of new_cluster_size

// ---------------------------------------------------------------------------
// Kernel 1: per-code half-norms + pre-scaled EMA init of outputs.
// One wave per code (block = 8 waves = 8 codes).
// ---------------------------------------------------------------------------
__global__ __launch_bounds__(256)
void k_init(const float* __restrict__ embed,
            const float* __restrict__ cluster_size,
            const float* __restrict__ embed_avg,
            float* __restrict__ out, float* __restrict__ ws) {
  const int lane = threadIdx.x & 31;
  const int wid  = threadIdx.x >> 5;
  const int code = blockIdx.x * 8 + wid;
  float ss = 0.f;
#pragma unroll
  for (int i = 0; i < 8; ++i) {
    const int d = lane + i * 32;
    const float e = embed[(long)code * D + d];
    ss += e * e;
    // new_embed_avg starts at decay*embed_avg; segment-sum atomics add on top
    out[OFF_NEA + (long)code * D + d] = embed_avg[(long)code * D + d] * DECAY;
  }
#pragma unroll
  for (int off = 16; off > 0; off >>= 1) ss += __shfl_xor(ss, off, 32);
  if (lane == 0) {
    ws[WS_HN + code]     = 0.5f * ss;
    out[OFF_NCS + code]  = cluster_size[code] * DECAY;  // counts add on top
  }
  if (blockIdx.x == 0 && threadIdx.x == 0) ws[WS_TOTAL] = 0.f;
}

// ---------------------------------------------------------------------------
// Kernel 2: WMMA nearest-code search.
// argmin_k (0.5*||e_k||^2 - x.e_k)  ==  argmin_k dist(x, e_k)
// 8 waves/WG; each wave owns TWO 16-row A strips (32 rows) so every B
// fragment loaded from LDS feeds two v_wmma_f32_16x16x32_bf16 issues.
// Embed staged in LDS as bf16 in 128-code blocks.
// ---------------------------------------------------------------------------
constexpr int KB   = 128;     // codes per LDS block (64 KB of bf16)
constexpr int NBLK = K / KB;  // 16 blocks
constexpr int ROWS_PER_WG = 256;  // 8 waves * 32 rows

__global__ __launch_bounds__(256)
void k_argmin(const float* __restrict__ x,
              const float* __restrict__ embed,
              const float* __restrict__ halfnorm,
              float* __restrict__ out_ind) {
  __shared__ alignas(32) __bf16 sB[KB * D];   // [code][d] row-major, 64 KB

  const int tid  = threadIdx.x;
  const int lane = tid & 31;
  const int m    = lane & 15;   // row slot (A) / code slot (C)
  const int hi   = lane >> 4;   // lane half
  const int wid  = tid >> 5;
  const int rowBase = blockIdx.x * ROWS_PER_WG + wid * 32;

  // ---- A fragments: 16x32 bf16 per k-tile, 8 tiles cover D=256.
  // lanes 0-15 hold row M=lane with K = {0..15};
  // lanes 16-31 hold same rows with K = {16..31} (matches B).
  v16bf a0[8], a1[8];
  const int kb = hi * 8;
  const float* __restrict__ xr0 = x + (long)(rowBase + m) * D;
  const float* __restrict__ xr1 = x + (long)(rowBase + 16 + m) * D;
#pragma unroll
  for (int t = 0; t < 8; ++t) {
#pragma unroll
    for (int j = 0; j < 16; ++j) {
      const int k = t * 32 + hi * 16 + j;
      a0[t][j] = (__bf16)xr0[k];
      a1[t][j] = (__bf16)xr1[k];
    }
  }

  float rmin0[8], ridx0[8], rmin1[8], ridx1[8];
#pragma unroll
  for (int r = 0; r < 8; ++r) {
    rmin0[r] = 3.4e38f; ridx0[r] = 0.f;
    rmin1[r] = 3.4e38f; ridx1[r] = 0.f;
  }

  for (int blk = 0; blk < NBLK; ++blk) {
    const int cb = blk * KB;
    __syncthreads();
    // cooperative stage: f32 global -> bf16 LDS (float4 granules)
    for (int it = 0; it < (KB * D / 4) / 256; ++it) {
      const int g    = tid + it * 256;       // float4-group id
      const int code = g >> 6;               // 64 groups per 256-wide row
      const int dpos = (g & 63) * 4;
      const float4 v = *reinterpret_cast<const float4*>(
          &embed[(long)(cb + code) * D + dpos]);
      v4bf b;
      b[0] = (__bf16)v.x; b[1] = (__bf16)v.y;
      b[2] = (__bf16)v.z; b[3] = (__bf16)v.w;
      *reinterpret_cast<v4bf*>(&sB[code * D + dpos]) = b;
    }
    __syncthreads();

    for (int ct = 0; ct < KB / 16; ++ct) {
      const int cLocal = ct * 16 + m;        // this lane's code column
      const int codeN  = cb + cLocal;
      v8f acc0 = {}, acc1 = {};
#pragma unroll
      for (int t = 0; t < 8; ++t) {
        // B 32x16: lane = column N, lane-half selects K 0..15 / 16..31
        // -> 16 contiguous bf16 (32B aligned) from this code's row.
        const v16bf b = *reinterpret_cast<const v16bf*>(
            &sB[cLocal * D + t * 32 + hi * 16]);
        acc0 = __builtin_amdgcn_wmma_f32_16x16x32_bf16(
            false, a0[t], false, b, (short)0, acc0, false, false);
        acc1 = __builtin_amdgcn_wmma_f32_16x16x32_bf16(
            false, a1[t], false, b, (short)0, acc1, false, false);
      }
      const float cn   = halfnorm[codeN];
      const float cidx = (float)codeN;
#pragma unroll
      for (int r = 0; r < 8; ++r) {
        const float s0 = cn - acc0[r];       // 0.5||e||^2 - x.e
        if (s0 < rmin0[r]) { rmin0[r] = s0; ridx0[r] = cidx; }
        const float s1 = cn - acc1[r];
        if (s1 < rmin1[r]) { rmin1[r] = s1; ridx1[r] = cidx; }
      }
    }
  }

  // argmin across the 16 code slots (lanes within each half)
#pragma unroll
  for (int off = 8; off > 0; off >>= 1) {
#pragma unroll
    for (int r = 0; r < 8; ++r) {
      float omin = __shfl_xor(rmin0[r], off, 16);
      float oidx = __shfl_xor(ridx0[r], off, 16);
      if (omin < rmin0[r] || (omin == rmin0[r] && oidx < ridx0[r])) {
        rmin0[r] = omin; ridx0[r] = oidx;
      }
      omin = __shfl_xor(rmin1[r], off, 16);
      oidx = __shfl_xor(ridx1[r], off, 16);
      if (omin < rmin1[r] || (omin == rmin1[r] && oidx < ridx1[r])) {
        rmin1[r] = omin; ridx1[r] = oidx;
      }
    }
  }
  // C layout: reg r holds row r (lanes 0-15) / row 8+r (lanes 16-31)
  if (m == 0) {
#pragma unroll
    for (int r = 0; r < 8; ++r) {
      out_ind[rowBase + hi * 8 + r]      = ridx0[r];
      out_ind[rowBase + 16 + hi * 8 + r] = ridx1[r];
    }
  }
}

// ---------------------------------------------------------------------------
// Kernel 3: gather quantize rows + fp32-atomic segment-sum into EMA outputs.
// One wave per row.
// ---------------------------------------------------------------------------
__global__ __launch_bounds__(256)
void k_gather(const float* __restrict__ x,
              const float* __restrict__ embed,
              float* __restrict__ out) {
  const int lane = threadIdx.x & 31;
  const int wid  = threadIdx.x >> 5;
  const int row  = blockIdx.x * 8 + wid;
  const int k    = (int)out[OFF_IND + row];
  const float* __restrict__ er = embed + (long)k * D;
  const float* __restrict__ xr = x + (long)row * D;
  float* __restrict__ qr  = out + OFF_Q + (long)row * D;
  float* __restrict__ nea = out + OFF_NEA + (long)k * D;
#pragma unroll
  for (int i = 0; i < 2; ++i) {
    const int d = lane * 4 + i * 128;
    const float4 e = *reinterpret_cast<const float4*>(&er[d]);
    *reinterpret_cast<float4*>(&qr[d]) = e;
    const float4 xv = *reinterpret_cast<const float4*>(&xr[d]);
    atomicAdd(&nea[d + 0], ONE_MINUS_DECAY * xv.x);
    atomicAdd(&nea[d + 1], ONE_MINUS_DECAY * xv.y);
    atomicAdd(&nea[d + 2], ONE_MINUS_DECAY * xv.z);
    atomicAdd(&nea[d + 3], ONE_MINUS_DECAY * xv.w);
  }
  if (lane == 0) atomicAdd(out + OFF_NCS + k, ONE_MINUS_DECAY);
}

// ---------------------------------------------------------------------------
// Kernel 4: total = sum(new_cluster_size). Single 1024-thread block.
// ---------------------------------------------------------------------------
__global__ __launch_bounds__(1024)
void k_total(const float* __restrict__ ncs, float* __restrict__ ws) {
  __shared__ float s[1024];
  s[threadIdx.x] = ncs[threadIdx.x] + ncs[threadIdx.x + 1024];
  __syncthreads();
  for (int off = 512; off > 0; off >>= 1) {
    if (threadIdx.x < (unsigned)off) s[threadIdx.x] += s[threadIdx.x + off];
    __syncthreads();
  }
  if (threadIdx.x == 0) ws[WS_TOTAL] = s[0];
}

// ---------------------------------------------------------------------------
// Kernel 5: laplace smoothing + new_embed = new_embed_avg / smoothed[k].
// ---------------------------------------------------------------------------
__global__ __launch_bounds__(256)
void k_final(float* __restrict__ out, const float* __restrict__ ws) {
  const long tid = (long)blockIdx.x * 256 + threadIdx.x;  // over K*D
  const int  k   = (int)(tid >> 8);
  const float total    = ws[WS_TOTAL];
  const float ncs      = out[OFF_NCS + k];
  const float smoothed = (ncs + EPSF) / (total + EPSF * (float)K) * total;
  out[OFF_NE + tid] = out[OFF_NEA + tid] / smoothed;
}

// ---------------------------------------------------------------------------
extern "C" void kernel_launch(void* const* d_in, const int* in_sizes, int n_in,
                              void* d_out, int out_size, void* d_ws, size_t ws_size,
                              hipStream_t stream) {
  const float* x   = (const float*)d_in[0];   // [N, D]
  const float* emb = (const float*)d_in[1];   // [K, D]
  const float* cs  = (const float*)d_in[2];   // [K]
  const float* ea  = (const float*)d_in[3];   // [K, D]
  float* out = (float*)d_out;
  float* ws  = (float*)d_ws;

  k_init  <<<K / 8,           256,  0, stream>>>(emb, cs, ea, out, ws);
  k_argmin<<<N / ROWS_PER_WG, 256,  0, stream>>>(x, emb, ws + WS_HN, out + OFF_IND);
  k_gather<<<N / 8,           256,  0, stream>>>(x, emb, out);
  k_total <<<1,               1024, 0, stream>>>(out + OFF_NCS, ws);
  k_final <<<(K * D) / 256,   256,  0, stream>>>(out, ws);
}
